// LossFunction_73443940761668
// MI455X (gfx1250) — matrix operand order repaired and checked by reading, not compiled, not run on hardware
//
#include <hip/hip_runtime.h>
#include <hip/hip_bf16.h>
#include <math.h>

typedef __bf16 bf16_t;
typedef __attribute__((ext_vector_type(16))) __bf16 v16bf;
typedef __attribute__((ext_vector_type(8)))  __bf16 v8bf;
typedef __attribute__((ext_vector_type(8)))  float  v8f;

union V16 { v16bf v; v8bf h[2]; };

#define N_ROWS   1024
#define DIM      512
#define NCLS     100000
#define NCLS_PAD 100096     // 391 * 256
#define TILE_M   64
#define TILE_N   256
#define NCB      391        // ceil(NCLS / TILE_N)
#define RB       16         // N_ROWS / TILE_M
#define T_ALPHA  0.98f
#define EPSC     0.001f

static __device__ __forceinline__ float neg_inf() { return -__builtin_inff(); }

// ---------------- kernel 1: per-row norms, normalized x -> bf16 -------------
__global__ void k_rowstats(const float* __restrict__ x, bf16_t* __restrict__ xn,
                           float* __restrict__ safe_norms) {
  int row = blockIdx.x;
  int t   = threadIdx.x;               // 128 threads
  __shared__ float red[128];
  const float* xr = x + (size_t)row * DIM;
  float v[4]; float s = 0.f;
#pragma unroll
  for (int i = 0; i < 4; ++i) { v[i] = xr[t + i * 128]; s += v[i] * v[i]; }
  red[t] = s; __syncthreads();
  for (int o = 64; o > 0; o >>= 1) { if (t < o) red[t] += red[t + o]; __syncthreads(); }
  float nrm = sqrtf(red[0]);
  float inv = 1.0f / fmaxf(nrm, 1e-12f);
#pragma unroll
  for (int i = 0; i < 4; ++i) xn[(size_t)row * DIM + t + i * 128] = (bf16_t)(v[i] * inv);
  if (t == 0) safe_norms[row] = fminf(fmaxf(nrm, 0.001f), 200.0f);
}

// ---------------- kernel 2: per-class norms, normalized W -> bf16 -----------
__global__ void k_wnorm(const float* __restrict__ w, bf16_t* __restrict__ wn) {
  int col = blockIdx.x;                // 0..NCLS_PAD-1
  int t   = threadIdx.x;               // 128 threads
  __shared__ float red[128];
  if (col >= NCLS) {                   // zero-fill pad columns (uniform branch)
#pragma unroll
    for (int i = 0; i < 4; ++i) wn[(size_t)col * DIM + t + i * 128] = (bf16_t)0.0f;
    return;
  }
  const float* wr = w + (size_t)col * DIM;
  float v[4]; float s = 0.f;
#pragma unroll
  for (int i = 0; i < 4; ++i) { v[i] = wr[t + i * 128]; s += v[i] * v[i]; }
  red[t] = s; __syncthreads();
  for (int o = 64; o > 0; o >>= 1) { if (t < o) red[t] += red[t + o]; __syncthreads(); }
  float inv = 1.0f / fmaxf(sqrtf(red[0]), 1e-12f);
#pragma unroll
  for (int i = 0; i < 4; ++i) wn[(size_t)col * DIM + t + i * 128] = (bf16_t)(v[i] * inv);
}

// ---------------- kernel 3: margin scaler per row ---------------------------
__global__ void k_margin(const float* __restrict__ safe_norms,
                         const float* __restrict__ batch_mean,
                         float* __restrict__ margin) {
  int t = threadIdx.x;                 // 256 threads, 1 block
  __shared__ float red[256];
  __shared__ float nbm_s;
  float s = 0.f;
#pragma unroll
  for (int i = 0; i < 4; ++i) s += safe_norms[t + i * 256];
  red[t] = s; __syncthreads();
  for (int o = 128; o > 0; o >>= 1) { if (t < o) red[t] += red[t + o]; __syncthreads(); }
  if (t == 0) nbm_s = (red[0] / (float)N_ROWS) * T_ALPHA + (1.0f - T_ALPHA) * batch_mean[0];
  __syncthreads();
  float nbm = nbm_s;
#pragma unroll
  for (int i = 0; i < 4; ++i) {
    int r = t + i * 256;
    margin[r] = (safe_norms[r] > nbm) ? 1.0f : -1.0f;
  }
}

// ---------------- kernel 4: raw logit at the label column only --------------
__global__ void k_rawlabel(const float* __restrict__ x, const float* __restrict__ w,
                           const int* __restrict__ labels, float* __restrict__ rawl) {
  int row = blockIdx.x;
  int t   = threadIdx.x;               // 128 threads
  __shared__ float red[128];
  int lab = labels[row];
  const float* xr = x + (size_t)row * DIM;
  const float* wr = w + (size_t)lab * DIM;
  float s = 0.f;
#pragma unroll
  for (int i = 0; i < 4; ++i) { int k = t + i * 128; s += xr[k] * wr[k]; }
  red[t] = s; __syncthreads();
  for (int o = 64; o > 0; o >>= 1) { if (t < o) red[t] += red[t + o]; __syncthreads(); }
  if (t == 0) rawl[row] = red[0];
}

// ---------------- kernel 5: fused WMMA GEMM + online softmax epilogue -------
// grid = (NCB, RB), block = 256 (8 waves). Tile: 64 rows x 256 cols, K = 512.
// Each wave: 32 rows x 64 cols = 2x4 WMMA tiles -> 8 WMMAs per 32-K step from
// 2 A-fragments + 4 B-fragments (high math-per-load for latency hiding).
__global__ __launch_bounds__(256)
void k_gemm(const bf16_t* __restrict__ xn, const bf16_t* __restrict__ wn,
            const int* __restrict__ labels, const float* __restrict__ rawl,
            float* __restrict__ gmax, float* __restrict__ gsum,
            float* __restrict__ gcos, int* __restrict__ gidx) {
  __shared__ bf16_t As[TILE_M][128];        // 16 KB A chunk
  __shared__ float  smax[TILE_M][4], ssum[TILE_M][4], scos[TILE_M][4];
  __shared__ int    sidx[TILE_M][4];

  const int cb = blockIdx.x, rb = blockIdx.y;
  const int tid  = threadIdx.x;
  const int wave = tid >> 5, lane = tid & 31;
  const int wm   = wave & 1;          // row group (2 x 32 rows)
  const int wnid = wave >> 1;         // col group (4 x 64 cols)
  const int hi   = lane >> 4, ln = lane & 15;
  const int rowbase = rb * TILE_M;
  const int colbase = cb * TILE_N;

  v8f zero8 = {0.f,0.f,0.f,0.f,0.f,0.f,0.f,0.f};
  v8f acc[2][4];
#pragma unroll
  for (int i = 0; i < 2; ++i)
#pragma unroll
    for (int j = 0; j < 4; ++j) acc[i][j] = zero8;

  // per-lane B base pointer: column = colbase + wnid*64 + tn*16 + ln
  const size_t brow0 = (size_t)(colbase + wnid * 64 + ln) * DIM;

  for (int ko = 0; ko < 4; ++ko) {          // 128-wide K chunks
    const int kbase = ko * 128;
    // stage 64x128 bf16 A chunk into LDS (16-byte vectors)
#pragma unroll
    for (int vv = 0; vv < 4; ++vv) {
      int e = (vv * 256 + tid) * 8;
      int r = e >> 7, k = e & 127;
      *(uint4*)&As[r][k] = *(const uint4*)(xn + (size_t)(rowbase + r) * DIM + kbase + k);
    }
    // warm GL2 for the next K chunk of the B stream (global_prefetch_b8)
    if (ko < 3) {
#pragma unroll
      for (int tn = 0; tn < 4; ++tn)
        __builtin_prefetch(wn + brow0 + (size_t)tn * 16 * DIM + kbase + 128 + hi * 16, 0, 1);
    }
    __syncthreads();
#pragma unroll
    for (int ks = 0; ks < 4; ++ks) {        // 32-deep WMMA K steps
      const int kc = ks * 32;
      V16 a0, a1, b[4];
      // A frags: lane halves carry K 0-15 (hi=0) and 16-31 (hi=1)
      {
        int r0 = wm * 32 + ln;
        a0.h[0] = *(const v8bf*)&As[r0][kc + hi * 16];
        a0.h[1] = *(const v8bf*)&As[r0][kc + hi * 16 + 8];
        int r1 = wm * 32 + 16 + ln;
        a1.h[0] = *(const v8bf*)&As[r1][kc + hi * 16];
        a1.h[1] = *(const v8bf*)&As[r1][kc + hi * 16 + 8];
      }
      // B frags: 32 contiguous bytes along K at K = 16*hi, lane = N
      {
        const size_t kg = (size_t)(kbase + kc + hi * 16);
#pragma unroll
        for (int tn = 0; tn < 4; ++tn) {
          const bf16_t* p = wn + brow0 + (size_t)tn * 16 * DIM + kg;
          b[tn].h[0] = *(const v8bf*)p;
          b[tn].h[1] = *(const v8bf*)(p + 8);
        }
      }
#pragma unroll
      for (int tn = 0; tn < 4; ++tn) {
        acc[0][tn] = __builtin_amdgcn_wmma_f32_16x16x32_bf16(false, a0.v, false, b[tn].v, (short)0, acc[0][tn], false, false);
        acc[1][tn] = __builtin_amdgcn_wmma_f32_16x16x32_bf16(false, a1.v, false, b[tn].v, (short)0, acc[1][tn], false, false);
      }
    }
    __syncthreads();
  }

  // ---- epilogue: clip -> cosine; label substitution; online stats ----------
  const float NI = neg_inf();
  for (int tm = 0; tm < 2; ++tm) {
    float mx[8], cs[8], se[8]; int mi[8]; int labv[8]; float rawv[8];
#pragma unroll
    for (int r = 0; r < 8; ++r) {
      mx[r] = NI; cs[r] = 0.f; se[r] = 0.f; mi[r] = 0x7fffffff;
      int rg = rowbase + wm * 32 + tm * 16 + hi * 8 + r;   // C layout: M = 8*hi + vgpr
      labv[r] = labels[rg]; rawv[r] = rawl[rg];
    }
#pragma unroll
    for (int tn = 0; tn < 4; ++tn) {
      int jc = colbase + wnid * 64 + tn * 16 + ln;
      bool inb = jc < NCLS;
#pragma unroll
      for (int r = 0; r < 8; ++r) {
        float c = acc[tm][tn][r];
        c = fminf(fmaxf(c, -(1.0f - EPSC)), 1.0f - EPSC);
        float v = (jc == labv[r]) ? rawv[r] : c;
        if (!inb) { c = 0.f; v = NI; }
        cs[r] += c;
        if (v > mx[r] || (v == mx[r] && jc < mi[r])) { mx[r] = v; mi[r] = jc; }
      }
    }
    // butterfly reduce over the 16 lanes sharing each row (width 16)
    for (int off = 1; off < 16; off <<= 1) {
#pragma unroll
      for (int r = 0; r < 8; ++r) {
        float om = __shfl_xor(mx[r], off, 16);
        int   oi = __shfl_xor(mi[r], off, 16);
        float oc = __shfl_xor(cs[r], off, 16);
        cs[r] += oc;
        if (om > mx[r] || (om == mx[r] && oi < mi[r])) { mx[r] = om; mi[r] = oi; }
      }
    }
    // sum of exp relative to the reduced row max
#pragma unroll
    for (int tn = 0; tn < 4; ++tn) {
      int jc = colbase + wnid * 64 + tn * 16 + ln;
      bool inb = jc < NCLS;
#pragma unroll
      for (int r = 0; r < 8; ++r) {
        if (!inb) continue;
        float c = acc[tm][tn][r];
        c = fminf(fmaxf(c, -(1.0f - EPSC)), 1.0f - EPSC);
        float v = (jc == labv[r]) ? rawv[r] : c;
        se[r] += __expf(v - mx[r]);
      }
    }
    for (int off = 1; off < 16; off <<= 1) {
#pragma unroll
      for (int r = 0; r < 8; ++r) se[r] += __shfl_xor(se[r], off, 16);
    }
    if (ln == 0) {
#pragma unroll
      for (int r = 0; r < 8; ++r) {
        int rl = wm * 32 + tm * 16 + hi * 8 + r;
        smax[rl][wnid] = mx[r]; ssum[rl][wnid] = se[r];
        scos[rl][wnid] = cs[r]; sidx[rl][wnid] = mi[r];
      }
    }
  }
  __syncthreads();
  // combine 4 col-wave partials per row, emit per-(row, col-block) partials
  if (tid < TILE_M) {
    float M = NI, S = 0.f, Ct = 0.f; int I = 0x7fffffff;
#pragma unroll
    for (int q = 0; q < 4; ++q) {
      float m2 = smax[tid][q];
      Ct += scos[tid][q];
      if (m2 > NI) {
        float s2 = ssum[tid][q]; int i2 = sidx[tid][q];
        if (m2 > M || (m2 == M && i2 < I)) I = i2;
        float nm = fmaxf(M, m2);
        S = S * __expf(M - nm) + s2 * __expf(m2 - nm);
        M = nm;
      }
    }
    size_t id = (size_t)(rowbase + tid) * NCB + cb;
    gmax[id] = M; gsum[id] = S; gcos[id] = Ct; gidx[id] = I;
  }
}

// ---------------- kernel 6: per-row merge across col-blocks -----------------
__global__ void k_rowreduce(const float* __restrict__ gmax, const float* __restrict__ gsum,
                            const float* __restrict__ gcos, const int* __restrict__ gidx,
                            const int* __restrict__ labels, const float* __restrict__ rawl,
                            const float* __restrict__ margin,
                            const int* __restrict__ ith, const int* __restrict__ tot,
                            float* __restrict__ rowloss, float* __restrict__ rowcorrect) {
  int row = blockIdx.x, t = threadIdx.x;    // 256 threads
  const float NI = neg_inf();
  float M = NI, S = 0.f, Ct = 0.f; int I = 0x7fffffff;
  for (int cb = t; cb < NCB; cb += 256) {
    size_t id = (size_t)row * NCB + cb;
    float m2 = gmax[id];
    Ct += gcos[id];
    if (m2 > NI) {
      float s2 = gsum[id]; int i2 = gidx[id];
      if (m2 > M || (m2 == M && i2 < I)) I = i2;
      float nm = fmaxf(M, m2);
      S = S * __expf(M - nm) + s2 * __expf(m2 - nm);
      M = nm;
    }
  }
  __shared__ float aM[256], aS[256], aC[256]; __shared__ int aI[256];
  aM[t] = M; aS[t] = S; aC[t] = Ct; aI[t] = I;
  __syncthreads();
  if (t == 0) {
    float M0 = aM[0], S0 = aS[0], C0 = aC[0]; int I0 = aI[0];
    for (int q = 1; q < 256; ++q) {
      float m2 = aM[q];
      C0 += aC[q];
      if (m2 > NI) {
        if (m2 > M0 || (m2 == M0 && aI[q] < I0)) I0 = aI[q];
        float nm = fmaxf(M0, m2);
        S0 = S0 * __expf(M0 - nm) + aS[q] * __expf(m2 - nm);
        M0 = nm;
      }
    }
    float logZ   = logf(S0) + M0;
    float ce     = logZ - rawl[row];                     // -log p(label)
    float lambda = (float)ith[0] / (float)tot[0];
    float Cf     = (float)NCLS;
    float Wrow   = lambda * (margin[row] * (Cf - C0) + 2.0f * Cf) + (1.0f - lambda) * Cf;
    rowloss[row]    = ce * Wrow;
    rowcorrect[row] = (I0 == labels[row]) ? 1.0f : 0.0f;
  }
}

// ---------------- kernel 7: deterministic final reduction -------------------
__global__ void k_final(const float* __restrict__ rowloss, const float* __restrict__ rowcorrect,
                        float* __restrict__ out) {
  int t = threadIdx.x;                      // 256 threads, 1 block
  __shared__ float rl[256], rc[256];
  float a = 0.f, b = 0.f;
  for (int i = t; i < N_ROWS; i += 256) { a += rowloss[i]; b += rowcorrect[i]; }
  rl[t] = a; rc[t] = b; __syncthreads();
  for (int o = 128; o > 0; o >>= 1) {
    if (t < o) { rl[t] += rl[t + o]; rc[t] += rc[t + o]; }
    __syncthreads();
  }
  if (t == 0) {
    out[0] = rl[0] / ((float)N_ROWS * (float)NCLS);
    out[1] = rc[0] * (100.0f / (float)N_ROWS);
  }
}

// ---------------- launcher --------------------------------------------------
extern "C" void kernel_launch(void* const* d_in, const int* in_sizes, int n_in,
                              void* d_out, int out_size, void* d_ws, size_t ws_size,
                              hipStream_t stream) {
  const float* x          = (const float*)d_in[0];
  const float* weight     = (const float*)d_in[1];
  const float* batch_mean = (const float*)d_in[2];
  const int*   labels     = (const int*)d_in[3];
  const int*   ith        = (const int*)d_in[4];
  const int*   tot        = (const int*)d_in[5];
  float*       out        = (float*)d_out;

  char* p = (char*)d_ws;
  bf16_t* wn = (bf16_t*)p;                 p += (size_t)NCLS_PAD * DIM * sizeof(bf16_t);
  bf16_t* xn = (bf16_t*)p;                 p += (size_t)N_ROWS * DIM * sizeof(bf16_t);
  float* safe_norms = (float*)p;           p += N_ROWS * sizeof(float);
  float* margin     = (float*)p;           p += N_ROWS * sizeof(float);
  float* rawl       = (float*)p;           p += N_ROWS * sizeof(float);
  float* rowloss    = (float*)p;           p += N_ROWS * sizeof(float);
  float* rowcorrect = (float*)p;           p += N_ROWS * sizeof(float);
  float* gmax = (float*)p;                 p += (size_t)N_ROWS * NCB * sizeof(float);
  float* gsum = (float*)p;                 p += (size_t)N_ROWS * NCB * sizeof(float);
  float* gcos = (float*)p;                 p += (size_t)N_ROWS * NCB * sizeof(float);
  int*   gidx = (int*)p;                   p += (size_t)N_ROWS * NCB * sizeof(int);

  k_rowstats<<<N_ROWS, 128, 0, stream>>>(x, xn, safe_norms);
  k_wnorm<<<NCLS_PAD, 128, 0, stream>>>(weight, wn);
  k_margin<<<1, 256, 0, stream>>>(safe_norms, batch_mean, margin);
  k_rawlabel<<<N_ROWS, 128, 0, stream>>>(x, weight, labels, rawl);
  dim3 g(NCB, RB);
  k_gemm<<<g, 256, 0, stream>>>(xn, wn, labels, rawl, gmax, gsum, gcos, gidx);
  k_rowreduce<<<N_ROWS, 256, 0, stream>>>(gmax, gsum, gcos, gidx, labels, rawl,
                                          margin, ith, tot, rowloss, rowcorrect);
  k_final<<<1, 256, 0, stream>>>(rowloss, rowcorrect, out);
}
